// JumpODE_39530878992478
// MI455X (gfx1250) — compile-verified, matrix operand order checked by reading, not run on hardware
//
#include <hip/hip_runtime.h>

typedef __attribute__((ext_vector_type(16))) _Float16 v16h;
typedef __attribute__((ext_vector_type(8)))  _Float16 v8h;
typedef __attribute__((ext_vector_type(8)))  float    v8f;

#define B_      2048
#define T_      128
#define H_      64
#define NSTEPS  10
#define MTILE   16
#define NTHREADS 128   // 4 wave32 waves

__device__ __forceinline__ int koff(int h) { return h + ((h >> 3) << 3); } // h + (h>=8 ? 8 : 0)

__device__ __forceinline__ v8f wmma_f16(v16h a, v16h b, v8f c) {
    return __builtin_amdgcn_wmma_f32_16x16x32_f16(false, a, false, b, (short)0, c, false, false);
}

// Combine two contiguous 8-half vectors into a v16h fragment register
__device__ __forceinline__ v16h combine(v8h lo, v8h hi) {
    v16h r;
#pragma unroll
    for (int h = 0; h < 8; ++h) { r[h] = lo[h]; r[h + 8] = hi[h]; }
    return r;
}

// A fragment (16x32 f16) from a row-major f16 LDS tile [16][ld] (ld multiple of 8).
// Pattern: halves 0..7 -> k = kb..kb+7 ; halves 8..15 -> k = kb+16..kb+23  => two b128 loads.
__device__ __forceinline__ v16h ldsA_f16(const _Float16* p, int ld, int lane, int kc) {
    int m  = lane & 15;
    int kb = ((lane >> 4) << 3) + (kc << 5);
    const _Float16* row = p + m * ld + kb;
    v8h lo = *(const v8h*)(row);
    v8h hi = *(const v8h*)(row + 16);
    return combine(lo, hi);
}

// A fragment from a row-major f32 LDS tile (setup only; scalar gather + cvt)
__device__ __forceinline__ v16h ldsA_f32(const float* p, int ld, int lane, int kc) {
    v16h a;
    int m  = lane & 15;
    int kb = ((lane >> 4) << 3) + (kc << 5);
    const float* row = p + m * ld + kb;
#pragma unroll
    for (int h = 0; h < 16; ++h) a[h] = (_Float16)row[koff(h)];
    return a;
}

// B fragment (32x16 f16), B[k][n] = W[n0+n][k0+k], W row-major in global f32 (setup only)
__device__ __forceinline__ v16h gatherB_g(const float* W, int ld, int n0, int k0, int lane) {
    v16h b;
    int n  = n0 + (lane & 15);
    int kb = k0 + ((lane >> 4) << 3);
    const float* row = W + n * ld + kb;
#pragma unroll
    for (int h = 0; h < 16; ++h) b[h] = (_Float16)row[koff(h)];
    return b;
}

// Load a pregathered fragment (32 contiguous bytes per lane) from LDS
__device__ __forceinline__ v16h loadFragLDS(const _Float16* p, int fragIdx, int lane) {
    const _Float16* q = p + (size_t)(fragIdx * 32 + lane) * 16;
    v8h lo = *(const v8h*)(q);
    v8h hi = *(const v8h*)(q + 8);
    return combine(lo, hi);
}

__device__ __forceinline__ float sigm(float x) { return 1.0f / (1.0f + __expf(-x)); }
__device__ __forceinline__ float softplus(float x) {
    return (x > 20.0f) ? x : log1pf(__expf(x));
}

__global__ void __launch_bounds__(NTHREADS)
jumpode_zero_ws(float* ws) {
    ws[0] = 0.0f;
    ws[1] = 0.0f;
}

__global__ void __launch_bounds__(NTHREADS)
jumpode_finalize(const float* ws, float* out) {
    out[0] = ws[0] / ws[1];
}

__global__ void __launch_bounds__(NTHREADS)
jumpode_main(const float* __restrict__ times, const int* __restrict__ marks,
             const float* __restrict__ mask,
             const float* __restrict__ W0, const float* __restrict__ b0,
             const float* __restrict__ W1, const float* __restrict__ b1,
             const float* __restrict__ W2, const float* __restrict__ b2,
             const float* __restrict__ Wi, const float* __restrict__ bi,
             const float* __restrict__ emb,
             const float* __restrict__ W_ih, const float* __restrict__ W_hh,
             const float* __restrict__ b_ih, const float* __restrict__ b_hh,
             float* __restrict__ out, float* __restrict__ ws)
{
    __shared__ alignas(16) float    hw[MTILE * H_];    // h_ode / carry state (f32)
    __shared__ alignas(16) _Float16 hwh[MTILE * H_];   // f16 mirror of hw (A-fragment source)
    __shared__ alignas(16) float    cst[MTILE * H_];   // LSTM cell state
    __shared__ alignas(16) _Float16 act1[MTILE * H_];  // MLP activation ping
    __shared__ alignas(16) _Float16 act2[MTILE * H_];  // MLP activation pong
    __shared__ alignas(16) float    m1s[MTILE * H_];   // mark embeddings (setup only)
    __shared__ alignas(16) float    gbase[MTILE * 256];// m1 @ W_ih[:,1:]^T + b_ih + b_hh
    __shared__ alignas(32) _Float16 whhFrag[32 * 32 * 16]; // 16 tiles x 2 kc, frag-major
    __shared__ float    w0c0[H_], b0s[H_], b1s[H_], b2s[H_], wis[H_];
    __shared__ float    wihc0[4 * H_];
    __shared__ float    tvec[MTILE];
    __shared__ float    integ[MTILE];
    __shared__ float    red[NTHREADS];
    __shared__ float    bis;

    const int tid   = threadIdx.x;
    const int lane  = tid & 31;
    const int w     = tid >> 5;            // wave id 0..3; owns columns [16w,16w+16)
    const int b0row = blockIdx.x * MTILE;
    const float dt  = 1.0f / (float)NSTEPS;

    // ---------------- setup ----------------
    for (int i = tid; i < H_; i += NTHREADS) {
        w0c0[i] = W0[i * (H_ + 1)];        // W0[:,0] (rank-1 column for s_t)
        b0s[i]  = b0[i];
        b1s[i]  = b1[i];
        b2s[i]  = b2[i];
        wis[i]  = Wi[i];
    }
    for (int i = tid; i < 4 * H_; i += NTHREADS)
        wihc0[i] = W_ih[i * (H_ + 1)];     // W_ih[:,0] (rank-1 column for t)
    if (tid == 0) bis = bi[0];
    for (int i = tid; i < MTILE * H_; i += NTHREADS) {
        hw[i] = 0.0f; cst[i] = 0.0f; hwh[i] = (_Float16)0.0f;
    }
    {   // m1 = emb[marks[:,1,0]] for this block's 16 rows
        int row = tid >> 3, cp = (tid & 7) * 8;
        int mk  = marks[(b0row + row) * T_ + 1];
#pragma unroll
        for (int k = 0; k < 8; ++k) m1s[row * H_ + cp + k] = emb[mk * H_ + cp + k];
    }
    // Pregather all W_hh B-fragments into LDS (fragment-major: 32B contiguous per lane).
    // fragIdx = nt*2 + kc, nt = gate-column tile 0..15.
    for (int idx = w; idx < 32; idx += 4) {
        int nt = idx >> 1, kc = idx & 1;
        v16h bf = gatherB_g(W_hh, H_, nt * 16, kc * 32, lane);
        _Float16* dst = &whhFrag[(size_t)(idx * 32 + lane) * 16];
        v8h lo, hi;
#pragma unroll
        for (int h = 0; h < 8; ++h) { lo[h] = bf[h]; hi[h] = bf[h + 8]; }
        *(v8h*)(dst)     = lo;
        *(v8h*)(dst + 8) = hi;
    }
    // persistent per-wave B fragments for the three MLP weight matrices
    v16h bW0[2], bW1[2], bW2[2];
#pragma unroll
    for (int kc = 0; kc < 2; ++kc) {
        bW0[kc] = gatherB_g(W0 + 1, H_ + 1, w * 16, kc * 32, lane); // W0[:,1:]
        bW1[kc] = gatherB_g(W1,     H_,     w * 16, kc * 32, lane);
        bW2[kc] = gatherB_g(W2,     H_,     w * 16, kc * 32, lane);
    }
    __syncthreads();

    // gate_base = m1 @ W_ih[:,1:]^T + b_ih + b_hh   (hoisted out of time loop)
    {
        v16h a0 = ldsA_f32(m1s, H_, lane, 0);
        v16h a1 = ldsA_f32(m1s, H_, lane, 1);
#pragma unroll
        for (int j = 0; j < 4; ++j) {
            int nt = w + 4 * j;
            v16h bb0 = gatherB_g(W_ih + 1, H_ + 1, nt * 16, 0,  lane);
            v16h bb1 = gatherB_g(W_ih + 1, H_ + 1, nt * 16, 32, lane);
            v8f acc = {};
            acc = wmma_f16(a0, bb0, acc);
            acc = wmma_f16(a1, bb1, acc);
            int n = nt * 16 + (lane & 15);
#pragma unroll
            for (int r = 0; r < 8; ++r) {
                int m = r + ((lane >> 4) << 3);
                gbase[m * 256 + n] = acc[r] + b_ih[n] + b_hh[n];
            }
        }
    }
    __syncthreads();

    float lossAcc = 0.0f, maskAcc = 0.0f;

    // ---------------- time loop ----------------
    for (int tt = 0; tt < T_; ++tt) {
        if (tid < MTILE) {
            tvec[tid]  = times[(size_t)(b0row + tid) * T_ + tt];
            integ[tid] = 0.0f;
        }
        __syncthreads();

        // ---- Euler integration: 10 sub-steps ----
        for (int s = 0; s < NSTEPS; ++s) {
            // intensity(hh) on the pre-update state: block-parallel dot with Wi
            {
                int row = tid >> 3, part = tid & 7;
                const float* hr = &hw[row * H_ + part * 8];
                float ps = 0.0f;
#pragma unroll
                for (int k = 0; k < 8; ++k) ps += hr[k] * wis[part * 8 + k];
                red[tid] = ps;
            }
            __syncthreads();
            if (tid < MTILE) {
                float sum = bis;
#pragma unroll
                for (int k = 0; k < 8; ++k) sum += red[tid * 8 + k];
                integ[tid] += dt * tvec[tid] * softplus(sum);
            }

            // GEMM1: act1 = tanh(hw @ W0[:,1:]^T + s_t * W0[:,0] + b0)
            {
                v16h a0 = ldsA_f16(hwh, H_, lane, 0);
                v16h a1 = ldsA_f16(hwh, H_, lane, 1);
                v8f acc = {};
                acc = wmma_f16(a0, bW0[0], acc);
                acc = wmma_f16(a1, bW0[1], acc);
                int n = w * 16 + (lane & 15);
                float wc = w0c0[n], bb = b0s[n];
#pragma unroll
                for (int r = 0; r < 8; ++r) {
                    int m = r + ((lane >> 4) << 3);
                    float st = (float)s * dt * tvec[m];
                    act1[m * H_ + n] = (_Float16)tanhf(acc[r] + wc * st + bb);
                }
            }
            __syncthreads();

            // GEMM2: act2 = tanh(act1 @ W1^T + b1)
            {
                v16h a0 = ldsA_f16(act1, H_, lane, 0);
                v16h a1 = ldsA_f16(act1, H_, lane, 1);
                v8f acc = {};
                acc = wmma_f16(a0, bW1[0], acc);
                acc = wmma_f16(a1, bW1[1], acc);
                int n = w * 16 + (lane & 15);
                float bb = b1s[n];
#pragma unroll
                for (int r = 0; r < 8; ++r) {
                    int m = r + ((lane >> 4) << 3);
                    act2[m * H_ + n] = (_Float16)tanhf(acc[r] + bb);
                }
            }
            __syncthreads();

            // GEMM3: hw += dt * t * tanh(act2 @ W2^T + b2); keep f16 mirror in sync
            {
                v16h a0 = ldsA_f16(act2, H_, lane, 0);
                v16h a1 = ldsA_f16(act2, H_, lane, 1);
                v8f acc = {};
                acc = wmma_f16(a0, bW2[0], acc);
                acc = wmma_f16(a1, bW2[1], acc);
                int n = w * 16 + (lane & 15);
                float bb = b2s[n];
#pragma unroll
                for (int r = 0; r < 8; ++r) {
                    int m = r + ((lane >> 4) << 3);
                    float nv = hw[m * H_ + n] + dt * tvec[m] * tanhf(acc[r] + bb);
                    hw[m * H_ + n]  = nv;
                    hwh[m * H_ + n] = (_Float16)nv;
                }
            }
            __syncthreads();
        }

        // ---- final intensity, NLL, loss accumulation ----
        {
            int row = tid >> 3, part = tid & 7;
            const float* hr = &hw[row * H_ + part * 8];
            float ps = 0.0f;
#pragma unroll
            for (int k = 0; k < 8; ++k) ps += hr[k] * wis[part * 8 + k];
            red[tid] = ps;
        }
        __syncthreads();
        if (tid < MTILE) {
            float sum = bis;
#pragma unroll
            for (int k = 0; k < 8; ++k) sum += red[tid * 8 + k];
            float inten = softplus(sum);
            float nll   = -__logf(inten) + integ[tid];
            float mk    = mask[(size_t)(b0row + tid) * T_ + tt];
            lossAcc += nll * mk;
            maskAcc += mk;
        }

        // ---- emit hidden = h_ode -> out[1 + (b*T + tt)*H + :] ----
        {
            int row = tid >> 3, col = (tid & 7) * 8;
            size_t base = 1 + ((size_t)(b0row + row) * T_ + tt) * H_ + col;
            const float* src = &hw[row * H_ + col];
#pragma unroll
            for (int k = 0; k < 8; ++k) out[base + k] = src[k];
        }

        // ---- LSTM update: gates = gbase + t*W_ih[:,0] + h_ode @ W_hh^T ----
        v16h ha0 = ldsA_f16(hwh, H_, lane, 0);
        v16h ha1 = ldsA_f16(hwh, H_, lane, 1);
        v8f accI = {}, accF = {}, accG = {}, accO = {};
        {
            // fragIdx = nt*2 + kc
            accI = wmma_f16(ha0, loadFragLDS(whhFrag, (w     ) * 2 + 0, lane), accI);
            accI = wmma_f16(ha1, loadFragLDS(whhFrag, (w     ) * 2 + 1, lane), accI);
            accF = wmma_f16(ha0, loadFragLDS(whhFrag, (w +  4) * 2 + 0, lane), accF);
            accF = wmma_f16(ha1, loadFragLDS(whhFrag, (w +  4) * 2 + 1, lane), accF);
            accG = wmma_f16(ha0, loadFragLDS(whhFrag, (w +  8) * 2 + 0, lane), accG);
            accG = wmma_f16(ha1, loadFragLDS(whhFrag, (w +  8) * 2 + 1, lane), accG);
            accO = wmma_f16(ha0, loadFragLDS(whhFrag, (w + 12) * 2 + 0, lane), accO);
            accO = wmma_f16(ha1, loadFragLDS(whhFrag, (w + 12) * 2 + 1, lane), accO);
        }
        __syncthreads();   // all reads of hw/hwh complete before overwriting

        {
            int n = w * 16 + (lane & 15);  // column within each 64-wide gate block
            float wci = wihc0[n], wcf = wihc0[n + 64];
            float wcg = wihc0[n + 128], wco = wihc0[n + 192];
#pragma unroll
            for (int r = 0; r < 8; ++r) {
                int m = r + ((lane >> 4) << 3);
                float tg = tvec[m];
                float gi = accI[r] + gbase[m * 256 + n]       + tg * wci;
                float gf = accF[r] + gbase[m * 256 + n + 64]  + tg * wcf;
                float gg = accG[r] + gbase[m * 256 + n + 128] + tg * wcg;
                float go = accO[r] + gbase[m * 256 + n + 192] + tg * wco;
                float cn = sigm(gf) * cst[m * H_ + n] + sigm(gi) * tanhf(gg);
                float hn = sigm(go) * tanhf(cn);
                cst[m * H_ + n] = cn;
                hw[m * H_ + n]  = hn;               // carry for next time step
                hwh[m * H_ + n] = (_Float16)hn;     // keep f16 mirror in sync
            }
        }
        __syncthreads();
    }

    // ---- loss reduction across blocks ----
    if (tid < MTILE) {
        atomicAdd(&ws[0], lossAcc);
        atomicAdd(&ws[1], maskAcc);
    }
}

extern "C" void kernel_launch(void* const* d_in, const int* in_sizes, int n_in,
                              void* d_out, int out_size, void* d_ws, size_t ws_size,
                              hipStream_t stream) {
    const float* times = (const float*)d_in[0];
    const int*   marks = (const int*)  d_in[1];
    const float* mask  = (const float*)d_in[2];
    const float* W0    = (const float*)d_in[3];
    const float* b0    = (const float*)d_in[4];
    const float* W1    = (const float*)d_in[5];
    const float* b1    = (const float*)d_in[6];
    const float* W2    = (const float*)d_in[7];
    const float* b2    = (const float*)d_in[8];
    const float* Wi    = (const float*)d_in[9];
    const float* bi    = (const float*)d_in[10];
    const float* emb   = (const float*)d_in[11];
    const float* W_ih  = (const float*)d_in[12];
    const float* W_hh  = (const float*)d_in[13];
    const float* b_ih  = (const float*)d_in[14];
    const float* b_hh  = (const float*)d_in[15];
    float* out = (float*)d_out;
    float* ws  = (float*)d_ws;

    jumpode_zero_ws<<<1, NTHREADS, 0, stream>>>(ws);
    jumpode_main<<<B_ / MTILE, NTHREADS, 0, stream>>>(
        times, marks, mask, W0, b0, W1, b1, W2, b2, Wi, bi,
        emb, W_ih, W_hh, b_ih, b_hh, out, ws);
    jumpode_finalize<<<1, NTHREADS, 0, stream>>>(ws, out);
}
